// GCN_62852551410182
// MI455X (gfx1250) — compile-verified, matrix operand order checked by reading, not run on hardware
//
#include <hip/hip_runtime.h>

// ---------------------------------------------------------------------------
// GCN (3-layer) for MI455X / gfx1250.
// Memory-bound on edge scatter (~19 GB @ 23.3 TB/s ~ 0.8 ms); GEMMs run on
// WMMA f16->f32 (16x16x32), weights pre-packed to the per-lane fragment order.
// ---------------------------------------------------------------------------

typedef __attribute__((ext_vector_type(16))) _Float16 v16h;
typedef __attribute__((ext_vector_type(8)))  float    v8f;

union AFrag { v16h v; struct { uint4 lo, hi; } u; };

#define HIDDEN 256

// ---------------- degree / normalization ----------------
__global__ void GCN_init_deg(float* __restrict__ deg, int N) {
    int i = blockIdx.x * blockDim.x + threadIdx.x;
    if (i < N) deg[i] = 1.0f;                 // self-loop contributes 1
}

__global__ void GCN_deg_acc(const long long* __restrict__ ei,
                            float* __restrict__ deg, long long E) {
    long long e = (long long)blockIdx.x * blockDim.x + threadIdx.x;
    if (e < E) atomicAdd(deg + ei[E + e], 1.0f);   // dst row of [2,E]
}

__global__ void GCN_rsqrt(float* __restrict__ d, int N) {
    int i = blockIdx.x * blockDim.x + threadIdx.x;
    if (i < N) d[i] = rsqrtf(d[i]);           // deg >= 1 always
}

// ---------------- weight packing into WMMA B-fragment order ----------------
// dst index = ((ntile*8 + kstep)*32 + lane)*16 + j
//   n = ntile*16 + lane%16 ; half = lane/16
//   k = kstep*32 + (j<8 ? 8*half + j : 16 + 8*half + (j-8))
__global__ void GCN_pack_w(const float* __restrict__ W, _Float16* __restrict__ Wp,
                           int NoutSrc, int NoutPad) {
    int idx = blockIdx.x * blockDim.x + threadIdx.x;
    if (idx >= NoutPad * 256) return;
    int j     = idx & 15;
    int lane  = (idx >> 4) & 31;
    int kstep = (idx >> 9) & 7;
    int ntile = idx >> 12;
    int half  = lane >> 4;
    int n = ntile * 16 + (lane & 15);
    int k = kstep * 32 + ((j < 8) ? (half * 8 + j) : (16 + half * 8 + (j - 8)));
    float v = (n < NoutSrc) ? W[k * NoutSrc + n] : 0.0f;
    Wp[idx] = (_Float16)v;
}

// ---------------- layer 1 (K=3, scalar; memory bound, trivial FLOPs) -------
__global__ void GCN_l1(const float* __restrict__ x, const float* __restrict__ W1,
                       const float* __restrict__ dinv, float* __restrict__ hs) {
    int idx = blockIdx.x * blockDim.x + threadIdx.x;     // n*256 + f
    int n = idx >> 8, f = idx & 255;
    float a = x[n * 3 + 0] * W1[f]
            + x[n * 3 + 1] * W1[256 + f]
            + x[n * 3 + 2] * W1[512 + f];
    hs[idx] = a * dinv[n];
}

// ---------------- WMMA GEMM: hs[N,NOUT] = (X[N,256] @ Wp) * dinv -----------
template <int NOUT>
__global__ void GCN_gemm_wmma(const _Float16* __restrict__ X,
                              const _Float16* __restrict__ Wp,
                              const float* __restrict__ dinv,
                              float* __restrict__ hs) {
    constexpr int NTILES = NOUT / 16;
    int lane = threadIdx.x & 31;
    int wave = threadIdx.x >> 5;
    int tile = blockIdx.x * (blockDim.x >> 5) + wave;    // exact tile count
    int mtile = tile / NTILES;
    int ntile = tile - mtile * NTILES;
    int m0   = mtile << 4;
    int half = lane >> 4;
    int mr   = lane & 15;

    const _Float16* xrow = X + (size_t)(m0 + mr) * 256;
    v8f c = {};
#pragma unroll
    for (int kb = 0; kb < 8; ++kb) {
        AFrag a;
        const _Float16* base = xrow + kb * 32;
        a.u.lo = *(const uint4*)(base + half * 8);        // K = 8h .. 8h+7
        a.u.hi = *(const uint4*)(base + 16 + half * 8);   // K = 16+8h .. +7
        v16h b = *(const v16h*)(Wp + (((size_t)(ntile * 8 + kb) * 32 + lane) << 4));
        c = __builtin_amdgcn_wmma_f32_16x16x32_f16(
                false, a.v, false, b, (short)0, c, false, false);
    }
    int n = (ntile << 4) + mr;
#pragma unroll
    for (int r = 0; r < 8; ++r) {
        int m = m0 + r + (half << 3);                     // C/D layout row
        hs[(size_t)m * NOUT + n] = c[r] * dinv[m];
    }
}

// ---------------- edge scatter: acc[dst] += hs[src] (f32 atomics) ----------
template <int F>
__global__ void GCN_scatter(const long long* __restrict__ ei,
                            const float* __restrict__ hs,
                            float* __restrict__ acc, long long E) {
    constexpr int F4 = F / 4;
    long long t = (long long)blockIdx.x * blockDim.x + threadIdx.x;
    if (t >= E * (long long)F4) return;
    long long e = t / F4;
    int c = (int)(t - e * F4);
    long long s = ei[e];
    long long d = ei[E + e];
    float4 v = *(const float4*)(hs + s * (long long)F + 4 * c);
    float* p = acc + d * (long long)F + 4 * c;
    atomicAdd(p + 0, v.x);
    atomicAdd(p + 1, v.y);
    atomicAdd(p + 2, v.z);
    atomicAdd(p + 3, v.w);
}

// ---------------- finalize: relu(dinv*(acc+hs)+b) -> f16 activations -------
__global__ void GCN_finalize_h(const float* __restrict__ acc,
                               const float* __restrict__ hs,
                               const float* __restrict__ dinv,
                               const float* __restrict__ b,
                               _Float16* __restrict__ Xh) {
    int idx = blockIdx.x * blockDim.x + threadIdx.x;     // n*256 + f
    int n = idx >> 8, f = idx & 255;
    float v = dinv[n] * (acc[idx] + hs[idx]) + b[f];     // self-loop = +hs
    Xh[idx] = (_Float16)fmaxf(v, 0.0f);
}

// ---------------- last layer: write f32 output (F padded 64 -> 50) ---------
__global__ void GCN_finalize_out(const float* __restrict__ acc,
                                 const float* __restrict__ hs,
                                 const float* __restrict__ dinv,
                                 const float* __restrict__ b3,
                                 float* __restrict__ out) {
    int idx = blockIdx.x * blockDim.x + threadIdx.x;     // n*64 + f
    int n = idx >> 6, f = idx & 63;
    if (f < 50) {
        float v = dinv[n] * (acc[idx] + hs[idx]) + b3[f];
        out[n * 50 + f] = fmaxf(v, 0.0f);
    }
}

// ---------------------------------------------------------------------------
extern "C" void kernel_launch(void* const* d_in, const int* in_sizes, int n_in,
                              void* d_out, int out_size, void* d_ws, size_t ws_size,
                              hipStream_t stream) {
    const float*     x   = (const float*)d_in[0];
    const long long* ei  = (const long long*)d_in[1];
    const float*     W1  = (const float*)d_in[2];
    const float*     b1  = (const float*)d_in[3];
    const float*     W2  = (const float*)d_in[4];
    const float*     b2  = (const float*)d_in[5];
    const float*     W3  = (const float*)d_in[6];
    const float*     b3  = (const float*)d_in[7];
    float*           out = (float*)d_out;

    const int       N = in_sizes[0] / 3;                 // 131072
    const long long E = (long long)in_sizes[1] / 2;      // 4194304

    // workspace layout (256B-aligned regions)
    char* ws = (char*)d_ws;
    const size_t o_dinv = 0;
    const size_t o_hs   = o_dinv + ((size_t)N * 4 + 255 & ~(size_t)255);
    const size_t o_acc  = o_hs   + (size_t)N * HIDDEN * 4;
    const size_t o_xh   = o_acc  + (size_t)N * HIDDEN * 4;
    const size_t o_w2p  = o_xh   + (size_t)N * HIDDEN * 2;
    const size_t o_w3p  = o_w2p  + (size_t)256 * 256 * 2;
    float*     dinv = (float*)(ws + o_dinv);
    float*     hs   = (float*)(ws + o_hs);
    float*     acc  = (float*)(ws + o_acc);
    _Float16*  Xh   = (_Float16*)(ws + o_xh);
    _Float16*  W2p  = (_Float16*)(ws + o_w2p);
    _Float16*  W3p  = (_Float16*)(ws + o_w3p);

    const int nodeBlk = N / 256;                         // 512
    const int nfBlk   = (N * HIDDEN) / 256;              // 131072

    // normalization coefficients (once)
    GCN_init_deg<<<nodeBlk, 256, 0, stream>>>(dinv, N);
    GCN_deg_acc<<<(unsigned)((E + 255) / 256), 256, 0, stream>>>(ei, dinv, E);
    GCN_rsqrt<<<nodeBlk, 256, 0, stream>>>(dinv, N);

    // pack weights into WMMA fragment order
    GCN_pack_w<<<(256 * 256) / 256, 256, 0, stream>>>(W2, W2p, 256, 256);
    GCN_pack_w<<<(256 * 64) / 256, 256, 0, stream>>>(W3, W3p, 50, 64);

    const unsigned sBlk256 = (unsigned)((E * 64 + 255) / 256);  // F=256 scatter
    const unsigned sBlk64  = (unsigned)((E * 16 + 255) / 256);  // F=64 scatter

    // ---- layer 1 (3 -> 256) ----
    GCN_l1<<<nfBlk, 256, 0, stream>>>(x, W1, dinv, hs);
    hipMemsetAsync(acc, 0, (size_t)N * HIDDEN * 4, stream);
    GCN_scatter<256><<<sBlk256, 256, 0, stream>>>(ei, hs, acc, E);
    GCN_finalize_h<<<nfBlk, 256, 0, stream>>>(acc, hs, dinv, b1, Xh);

    // ---- layer 2 (256 -> 256), WMMA ----
    GCN_gemm_wmma<256><<<(N / 16) * 16 / 8, 256, 0, stream>>>(Xh, W2p, dinv, hs);
    hipMemsetAsync(acc, 0, (size_t)N * HIDDEN * 4, stream);
    GCN_scatter<256><<<sBlk256, 256, 0, stream>>>(ei, hs, acc, E);
    GCN_finalize_h<<<nfBlk, 256, 0, stream>>>(acc, hs, dinv, b2, Xh);

    // ---- layer 3 (256 -> 50, padded to 64), WMMA ----
    GCN_gemm_wmma<64><<<(N / 16) * 4 / 8, 256, 0, stream>>>(Xh, W3p, dinv, hs);
    hipMemsetAsync(acc, 0, (size_t)N * 64 * 4, stream);
    GCN_scatter<64><<<sBlk64, 256, 0, stream>>>(ei, hs, acc, E);
    GCN_finalize_out<<<(N * 64) / 256, 256, 0, stream>>>(acc, hs, dinv, b3, out);
}
